// TreeLstmEncoderComplete_64501818851721
// MI455X (gfx1250) — matrix operand order repaired and checked by reading, not compiled, not run on hardware
//
#include <hip/hip_runtime.h>
#include <hip/hip_bf16.h>

// ---------------------------------------------------------------------------
// TreeLSTM VAE encoder for MI455X (gfx1250, wave32, WMMA).
// v_wmma_f32_16x16x32_f16 everywhere; B operands pre-packed to fragment
// layout (one v16h load each), A operands (h / embeddings) stored K-permuted
// (swap K bits 3<->4) so each A fragment is one v16h load. Branchless
// sigmoid/tanh via v_exp_f32 + v_rcp_f32. Levels 6..9 fused into a single
// workgroup with fence+barrier between levels. Child cell-state rows are
// prefetched (global_prefetch_b8) at tile entry to hide epilogue gathers.
// ---------------------------------------------------------------------------

typedef _Float16 v16h __attribute__((ext_vector_type(16)));
typedef float    v8f  __attribute__((ext_vector_type(8)));

#define TREES 32
#define PER   1023
#define NNODE (TREES * PER)
#define HID   128
#define EMB   128

__host__ __device__ __forceinline__ int permK(int k) {
    return (k & ~24) | ((k & 8) << 1) | ((k & 16) >> 1);
}

__device__ __forceinline__ v8f wmma_f16(v16h a, v16h b, v8f c) {
    return __builtin_amdgcn_wmma_f32_16x16x32_f16(
        false, a, false, b, (short)0, c, false, false);
}

// Branchless activations (v_exp_f32 / v_rcp_f32, no libm branches).
__device__ __forceinline__ float sigm_fast(float x) {
    return __builtin_amdgcn_rcpf(1.0f + __expf(-x));
}
__device__ __forceinline__ float tanh_fast(float x) {
    float xc = fminf(fmaxf(x, -10.0f), 10.0f);
    float e  = __expf(2.0f * xc);
    return (e - 1.0f) * __builtin_amdgcn_rcpf(e + 1.0f);
}

__device__ __forceinline__ v16h load_b_packed(const _Float16* __restrict__ P,
                                              int jtile, int KT, int kt, int half, int n) {
    return *(const v16h*)(P + ((((size_t)jtile * KT + kt) * 2 + half) * 16 + n) * 16);
}
__device__ __forceinline__ v16h load_a_perm(const _Float16* __restrict__ row,
                                            int kt, int half) {
    return *(const v16h*)(row + kt * 32 + half * 16);
}

// ----------------------- one-time layout transforms ------------------------
__global__ void pack_b_kernel(const float* __restrict__ src, _Float16* __restrict__ dst,
                              int nrows, int ncols) {
    int idx = blockIdx.x * blockDim.x + threadIdx.x;
    if (idx >= nrows * ncols) return;
    int KT = nrows >> 5;
    int e = idx & 15;
    int n = (idx >> 4) & 15;
    int half = (idx >> 8) & 1;
    int kt = (idx >> 9) % KT;
    int jtile = (idx >> 9) / KT;
    int K = kt * 32 + permK(half * 16 + e);
    dst[idx] = (_Float16)src[(size_t)K * ncols + jtile * 16 + n];
}

__global__ void perm_a_kernel(const float* __restrict__ src, _Float16* __restrict__ dst,
                              int count) {
    int i = blockIdx.x * blockDim.x + threadIdx.x;
    if (i < count) dst[i] = (_Float16)src[permK(i)];
}

// ------------------------------ leaf kernel --------------------------------
__global__ void leaf_kernel(const int* __restrict__ features,
                            const int* __restrict__ vocabs,
                            const _Float16* __restrict__ emb_leaf16,
                            const _Float16* __restrict__ Wleaf16,
                            const float* __restrict__ b_leaf,
                            _Float16* __restrict__ h16,
                            float* __restrict__ c32, int numTiles) {
    int wave = (blockIdx.x * blockDim.x + threadIdx.x) >> 5;
    if (wave >= numTiles) return;
    int lane = threadIdx.x & 31;
    int m = lane & 15, half = lane >> 4;

    int q = wave * 16 + m;
    int t = q >> 9;
    int loc = 511 + (q & 511);
    int node = t * PER + loc;
    int v = vocabs[node];
    int kv = v - 1; kv = kv < 0 ? 0 : (kv > 2 ? 2 : kv);
    const _Float16* xr = emb_leaf16 + ((size_t)kv * 5000 + features[node]) * 64;

    v16h ax[2];
#pragma unroll
    for (int kt = 0; kt < 2; ++kt) ax[kt] = load_a_perm(xr, kt, half);

    int vrow[8]; int nrow[8];
#pragma unroll
    for (int r = 0; r < 8; ++r) {
        int qr = wave * 16 + r + 8 * half;
        int tr = qr >> 9, lr = 511 + (qr & 511);
        nrow[r] = tr * PER + lr;
        vrow[r] = vocabs[nrow[r]];
    }
    int n = lane & 15;

#pragma unroll 1
    for (int j = 0; j < 8; ++j) {
        int col = j * 16 + n;
        int pcol = permK(col);
        float cn[8], hn[8];
#pragma unroll
        for (int r = 0; r < 8; ++r) { cn[r] = 0.0f; hn[r] = 0.0f; }
#pragma unroll 1
        for (int k = 0; k < 3; ++k) {
            const _Float16* Wk = Wleaf16 + (size_t)k * 64 * 512;
            v16h bi0 = load_b_packed(Wk, j,      2, 0, half, n);
            v16h bg0 = load_b_packed(Wk, 16 + j, 2, 0, half, n);
            v16h bo0 = load_b_packed(Wk, 24 + j, 2, 0, half, n);
            v16h bi1 = load_b_packed(Wk, j,      2, 1, half, n);
            v16h bg1 = load_b_packed(Wk, 16 + j, 2, 1, half, n);
            v16h bo1 = load_b_packed(Wk, 24 + j, 2, 1, half, n);
            v8f gi = {}, gg = {}, go = {};
            gi = wmma_f16(ax[0], bi0, gi); gi = wmma_f16(ax[1], bi1, gi);
            gg = wmma_f16(ax[0], bg0, gg); gg = wmma_f16(ax[1], bg1, gg);
            go = wmma_f16(ax[0], bo0, go); go = wmma_f16(ax[1], bo1, go);
            float bgi = b_leaf[k * 512 + col];
            float bgg = b_leaf[k * 512 + 256 + col];
            float bgo = b_leaf[k * 512 + 384 + col];
#pragma unroll
            for (int r = 0; r < 8; ++r) {
                float ck = sigm_fast(gi[r] + bgi) * tanh_fast(gg[r] + bgg);
                float hk = sigm_fast(go[r] + bgo) * tanh_fast(ck);
                cn[r] = (vrow[r] == k + 1) ? ck : cn[r];
                hn[r] = (vrow[r] == k + 1) ? hk : hn[r];
            }
        }
#pragma unroll
        for (int r = 0; r < 8; ++r) {
            c32[(size_t)nrow[r] * HID + col]  = cn[r];
            h16[(size_t)nrow[r] * HID + pcol] = (_Float16)hn[r];
        }
    }
}

// --------------------------- shared level tile -----------------------------
__device__ void level_tile(int tileIdx, int lc, int lane,
                           const int* __restrict__ features,
                           const _Float16* __restrict__ emb_res16,
                           const _Float16* __restrict__ Wiou16,
                           const _Float16* __restrict__ Uiou16,
                           const _Float16* __restrict__ Wf16,
                           const _Float16* __restrict__ Uf16,
                           const float* __restrict__ b_iou,
                           const float* __restrict__ b_f,
                           _Float16* __restrict__ h16,
                           float* __restrict__ c32) {
    int m = lane & 15, half = lane >> 4;
    int C = 1 << lc;

    int q = tileIdx * 16 + m;
    int t = q >> lc;
    int loc = (C - 1) + (q & (C - 1));
    int node = t * PER + loc;
    int c1 = t * PER + 2 * loc + 1;

    const _Float16* fr  = emb_res16 + (size_t)features[node] * EMB;
    const _Float16* h1r = h16 + (size_t)c1 * HID;
    const _Float16* h2r = h1r + HID;

    int prow[8], c1row[8];
#pragma unroll
    for (int r = 0; r < 8; ++r) {
        int qr = tileIdx * 16 + r + 8 * half;
        int tr = qr >> lc;
        int lr = (C - 1) + (qr & (C - 1));
        prow[r]  = tr * PER + lr;
        c1row[r] = tr * PER + 2 * lr + 1;
    }

    // Prefetch child c-rows (2*HID f32 = 1KB, contiguous): for rows 0-7 lanes 0-15
    // cover child 1 (512B), for rows 8-15 lanes 16-31 cover child 2; 32B stride -> lines hot before the
    // epilogue gathers below.
#pragma unroll
    for (int r = 0; r < 8; ++r)
        __builtin_prefetch(c32 + (size_t)c1row[r] * HID + lane * 8, 0, 0);

    v16h af[4], a1[4], a2[4], as[4];
#pragma unroll
    for (int kt = 0; kt < 4; ++kt) {
        af[kt] = load_a_perm(fr, kt, half);
        a1[kt] = load_a_perm(h1r, kt, half);
        a2[kt] = load_a_perm(h2r, kt, half);
        as[kt] = a1[kt] + a2[kt];
    }

    int n = lane & 15;

#pragma unroll 1
    for (int j = 0; j < 8; ++j) {
        v8f ai = {}, ao = {}, au = {}, awf = {}, au1 = {}, au2 = {};
#pragma unroll 2
        for (int kt = 0; kt < 4; ++kt) {
            v16h bWi = load_b_packed(Wiou16, j,      4, kt, half, n);
            v16h bUi = load_b_packed(Uiou16, j,      4, kt, half, n);
            v16h bWo = load_b_packed(Wiou16, 8 + j,  4, kt, half, n);
            v16h bUo = load_b_packed(Uiou16, 8 + j,  4, kt, half, n);
            v16h bWu = load_b_packed(Wiou16, 16 + j, 4, kt, half, n);
            v16h bUu = load_b_packed(Uiou16, 16 + j, 4, kt, half, n);
            v16h bWf = load_b_packed(Wf16, j, 4, kt, half, n);
            v16h bUf = load_b_packed(Uf16, j, 4, kt, half, n);
            ai  = wmma_f16(af[kt], bWi, ai);
            ai  = wmma_f16(as[kt], bUi, ai);
            ao  = wmma_f16(af[kt], bWo, ao);
            ao  = wmma_f16(as[kt], bUo, ao);
            au  = wmma_f16(af[kt], bWu, au);
            au  = wmma_f16(as[kt], bUu, au);
            awf = wmma_f16(af[kt], bWf, awf);
            au1 = wmma_f16(a1[kt], bUf, au1);
            au2 = wmma_f16(a2[kt], bUf, au2);
        }
        int col = j * 16 + n;
        int pcol = permK(col);
        float bi = b_iou[col];
        float bo = b_iou[128 + col];
        float bu = b_iou[256 + col];
        float bf = b_f[col];
#pragma unroll
        for (int r = 0; r < 8; ++r) {
            float f1 = sigm_fast(awf[r] + au1[r] + bf);
            float f2 = sigm_fast(awf[r] + au2[r] + bf);
            float cf = f1 * c32[(size_t)c1row[r] * HID + col]
                     + f2 * c32[((size_t)c1row[r] + 1) * HID + col];
            float cnew = sigm_fast(ai[r] + bi) * tanh_fast(au[r] + bu) + cf;
            float hnew = sigm_fast(ao[r] + bo) * tanh_fast(cnew);
            c32[(size_t)prow[r] * HID + col]  = cnew;
            h16[(size_t)prow[r] * HID + pcol] = (_Float16)hnew;
        }
    }
}

// --------------------------- wide level kernel -----------------------------
__global__ void level_kernel(const int* __restrict__ features,
                             const _Float16* __restrict__ emb_res16,
                             const _Float16* __restrict__ Wiou16,
                             const _Float16* __restrict__ Uiou16,
                             const _Float16* __restrict__ Wf16,
                             const _Float16* __restrict__ Uf16,
                             const float* __restrict__ b_iou,
                             const float* __restrict__ b_f,
                             _Float16* __restrict__ h16,
                             float* __restrict__ c32,
                             int lc, int numTiles) {
    int wave = (blockIdx.x * blockDim.x + threadIdx.x) >> 5;
    if (wave >= numTiles) return;
    level_tile(wave, lc, threadIdx.x & 31, features, emb_res16, Wiou16, Uiou16,
               Wf16, Uf16, b_iou, b_f, h16, c32);
}

// -------- fused tail: levels 6..9 in one workgroup (8 waves) ---------------
__global__ void tail_kernel(const int* __restrict__ features,
                            const _Float16* __restrict__ emb_res16,
                            const _Float16* __restrict__ Wiou16,
                            const _Float16* __restrict__ Uiou16,
                            const _Float16* __restrict__ Wf16,
                            const _Float16* __restrict__ Uf16,
                            const float* __restrict__ b_iou,
                            const float* __restrict__ b_f,
                            _Float16* __restrict__ h16,
                            float* __restrict__ c32) {
    int waveInBlk = threadIdx.x >> 5;  // 0..7
    int lane = threadIdx.x & 31;
#pragma unroll 1
    for (int nlev = 6; nlev <= 9; ++nlev) {
        int lc = 9 - nlev;
        int tiles = 2 << lc;  // 16, 8, 4, 2
        for (int tile = waveInBlk; tile < tiles; tile += 8)
            level_tile(tile, lc, lane, features, emb_res16, Wiou16, Uiou16,
                       Wf16, Uf16, b_iou, b_f, h16, c32);
        __threadfence();   // make h/c stores visible through L2
        __syncthreads();   // level barrier
    }
}

// ------------------------------ root kernel --------------------------------
__global__ void root_kernel(const _Float16* __restrict__ h16,
                            const float* __restrict__ Wm, const float* __restrict__ bm,
                            const float* __restrict__ Wv, const float* __restrict__ bv,
                            const float* __restrict__ eps, float* __restrict__ out) {
    int i = blockIdx.x * blockDim.x + threadIdx.x;
    if (i >= TREES * 64) return;
    int t = i >> 6, d = i & 63;
    const _Float16* hr = h16 + (size_t)t * PER * HID;
    float am = bm[d], av = bv[d];
#pragma unroll 4
    for (int k = 0; k < HID; ++k) {
        float hv = (float)hr[permK(k)];
        am += hv * Wm[k * 64 + d];
        av += hv * Wv[k * 64 + d];
    }
    float z = am + eps[i] * __expf(0.5f * av);
    out[i] = z;
    out[TREES * 64 + i] = am;
    out[2 * TREES * 64 + i] = av;
}

// ------------------------------- launcher ----------------------------------
extern "C" void kernel_launch(void* const* d_in, const int* in_sizes, int n_in,
                              void* d_out, int out_size, void* d_ws, size_t ws_size,
                              hipStream_t stream) {
    const int*   features = (const int*)d_in[0];
    const int*   vocabs   = (const int*)d_in[1];
    const float* eps      = (const float*)d_in[6];
    const float* emb_res  = (const float*)d_in[7];
    const float* emb_leaf = (const float*)d_in[8];
    const float* W_leaf   = (const float*)d_in[9];
    const float* b_leaf   = (const float*)d_in[10];
    const float* W_iou    = (const float*)d_in[11];
    const float* b_iou    = (const float*)d_in[12];
    const float* U_iou    = (const float*)d_in[13];
    const float* W_f      = (const float*)d_in[14];
    const float* b_f      = (const float*)d_in[15];
    const float* U_f      = (const float*)d_in[16];
    const float* Wm       = (const float*)d_in[17];
    const float* bm       = (const float*)d_in[18];
    const float* Wv       = (const float*)d_in[19];
    const float* bv       = (const float*)d_in[20];

    char* ws = (char*)d_ws;
    _Float16* h16 = (_Float16*)ws;                       // NNODE*128 f16 (A-permuted)
    float*    c32 = (float*)(ws + 8380416);              // NNODE*128 f32
    char* p = ws + 8380416 + 16760832;
    _Float16* Wiou16  = (_Float16*)p; p += 98304;
    _Float16* Uiou16  = (_Float16*)p; p += 98304;
    _Float16* Wf16    = (_Float16*)p; p += 32768;
    _Float16* Uf16    = (_Float16*)p; p += 32768;
    _Float16* Wleaf16 = (_Float16*)p; p += 196608;
    _Float16* embres16  = (_Float16*)p; p += 153600;
    _Float16* embleaf16 = (_Float16*)p; p += 1920000;

    pack_b_kernel<<<(49152 + 255) / 256, 256, 0, stream>>>(W_iou, Wiou16, 128, 384);
    pack_b_kernel<<<(49152 + 255) / 256, 256, 0, stream>>>(U_iou, Uiou16, 128, 384);
    pack_b_kernel<<<(16384 + 255) / 256, 256, 0, stream>>>(W_f, Wf16, 128, 128);
    pack_b_kernel<<<(16384 + 255) / 256, 256, 0, stream>>>(U_f, Uf16, 128, 128);
    for (int k = 0; k < 3; ++k)
        pack_b_kernel<<<(32768 + 255) / 256, 256, 0, stream>>>(
            W_leaf + (size_t)k * 64 * 512, Wleaf16 + (size_t)k * 64 * 512, 64, 512);
    perm_a_kernel<<<(76800 + 255) / 256, 256, 0, stream>>>(emb_res, embres16, 76800);
    perm_a_kernel<<<(960000 + 255) / 256, 256, 0, stream>>>(emb_leaf, embleaf16, 960000);

    // leaves: 1024 tiles, 4 waves/block
    leaf_kernel<<<256, 128, 0, stream>>>(features, vocabs, embleaf16, Wleaf16, b_leaf,
                                         h16, c32, 1024);

    // wide levels 1..5
    for (int nlev = 1; nlev <= 5; ++nlev) {
        int lc = 9 - nlev;
        int tiles = 2 << lc;
        int blocks = (tiles + 3) / 4;
        level_kernel<<<blocks, 128, 0, stream>>>(features, embres16, Wiou16, Uiou16,
                                                 Wf16, Uf16, b_iou, b_f, h16, c32,
                                                 lc, tiles);
    }

    // fused tail: levels 6..9 in one workgroup
    tail_kernel<<<1, 256, 0, stream>>>(features, embres16, Wiou16, Uiou16,
                                       Wf16, Uf16, b_iou, b_f, h16, c32);

    // roots
    root_kernel<<<16, 128, 0, stream>>>(h16, Wm, bm, Wv, bv, eps, (float*)d_out);
}
